// GNN_Model_27736898798406
// MI455X (gfx1250) — matrix operand order repaired and checked by reading, not compiled, not run on hardware
//
#include <hip/hip_runtime.h>
#include <hip/hip_bf16.h>
#include <math.h>

// ---- CDNA5 WMMA types (wave32) ----
typedef __attribute__((ext_vector_type(16))) __bf16 v16bf;
typedef __attribute__((ext_vector_type(8)))  float  v8f;

#define NF 16   // node features F
#define NH 16   // hidden H
#define NS 8    // edge features S
#define KPAD 160 // 128 (s*f) + 16 (bk rows) + 16 zero pad -> 5 k-steps of 32

__global__ __launch_bounds__(256) void zero_f32(float* p, int n) {
    int i = blockIdx.x * 256 + threadIdx.x;
    if (i < n) p[i] = 0.f;
}

// ---------------- Edge kernel: msg = o @ Wk2 (+ bk fold), scatter-add to agg --------------
__global__ __launch_bounds__(256) void ecc_edge_msg(
    const float* __restrict__ x, const float* __restrict__ e,
    const int* __restrict__ src, const int* __restrict__ dst,
    const float* __restrict__ Wk, const float* __restrict__ bk,
    float* __restrict__ agg, int N, int E)
{
    __shared__ __bf16 sWk[KPAD][NH];      // 5120 B, block-shared B operand
    __shared__ float  sX[8][16][NF];      // per-wave gathered x rows
    __shared__ float  sE[8][16][NS];      // per-wave edge features
    __shared__ int    sDst[8][16];

    const int tid  = threadIdx.x;
    const int wave = tid >> 5;
    const int lane = tid & 31;
    const int m    = lane & 15;           // row within tile / column n
    const bool hi  = lane >= 16;

    // Stage Wk2[k][h]: k = s*16+f -> Wk[s][f*16+h]; rows 128..143 = bk; 144..159 = 0
    for (int i = tid; i < KPAD * NH; i += 256) {
        int k = i >> 4, h = i & 15;
        float v;
        if (k < 128)      v = Wk[(k >> 4) * 256 + (k & 15) * 16 + h];
        else if (k < 144) v = bk[(k - 128) * 16 + h];
        else              v = 0.f;
        sWk[k][h] = (__bf16)v;
    }
    __syncthreads();

    const int numTiles = (E + 15) / 16;
    for (int tile = blockIdx.x * 8 + wave; tile < numTiles; tile += gridDim.x * 8) {
        const int ebase = tile * 16;
        // stage edge features (16x8)
        for (int i = lane; i < 16 * NS; i += 32) {
            int r = i >> 3, c = i & 7;
            int eid = ebase + r;
            sE[wave][r][c] = (eid < E) ? e[(size_t)eid * NS + c] : 0.f;
        }
        // stage dst
        if (lane < 16) {
            int eid = ebase + lane;
            sDst[wave][lane] = (eid < E) ? dst[eid] : -1;
        }
        // gather x[src] rows (16x16)
        for (int i = lane; i < 16 * NF; i += 32) {
            int r = i >> 4, c = i & 15;
            int eid = ebase + r;
            float v = 0.f;
            if (eid < E) v = x[(size_t)src[eid] * NF + c];
            sX[wave][r][c] = v;
        }
        // intra-wave LDS ops are in-order; no block barrier needed (per-wave staging)

        v8f acc = {};
        #pragma unroll
        for (int kt = 0; kt < 5; ++kt) {
            // A fragment: 16x32 bf16, contiguous k order shared with B.
            // lanes 0-15: K = kt*32 + 0..15; lanes 16-31: K = kt*32 + 16..31
            v16bf a;
            const int kA = kt * 32 + (hi ? 16 : 0);
            #pragma unroll
            for (int j = 0; j < 16; ++j) {
                int k = kA + j;
                float ov;
                if (k < 128)       ov = sE[wave][m][k >> 4] * sX[wave][m][k & 15]; // o = e_s * x_f
                else if (k < 144)  ov = sX[wave][m][k - 128];                      // bk fold
                else               ov = 0.f;                                        // pad
                a[j] = (__bf16)ov;
            }
            // B fragment: lane -> column n=m; lanes 0-15 K=kt*32+0..15, lanes 16-31 K=+16..31
            v16bf b;
            const int kB = kt * 32 + (hi ? 16 : 0);
            #pragma unroll
            for (int j = 0; j < 16; ++j) b[j] = sWk[kB + j][m];

            acc = __builtin_amdgcn_wmma_f32_16x16x32_bf16(
                false, a, false, b, (short)0, acc, false, false);
        }
        // C/D layout: VGPR r -> row (r or r+8), lane%16 -> h. Scatter-add.
        #pragma unroll
        for (int r = 0; r < 8; ++r) {
            int row = r + (hi ? 8 : 0);
            int d   = sDst[wave][row];
            if (d >= 0) atomicAdd(&agg[(size_t)d * NH + m], acc[r]);
        }
    }
}

// ---------------- Node kernel: h = BN(relu(agg + x@Wr + br)); per-block column max -------
__global__ __launch_bounds__(256) void node_update(
    const float* __restrict__ x, const float* __restrict__ agg,
    const float* __restrict__ Wr, const float* __restrict__ br,
    const float* __restrict__ gamma, const float* __restrict__ beta,
    const float* __restrict__ mmean, const float* __restrict__ mvar,
    float* __restrict__ blockMax, int N)
{
    __shared__ __bf16 sWr[32][NH];     // K padded 16 -> 32
    __shared__ float  sRed[8][NH];

    const int tid  = threadIdx.x;
    const int wave = tid >> 5;
    const int lane = tid & 31;
    const int m    = lane & 15;
    const bool hi  = lane >= 16;

    for (int i = tid; i < 32 * NH; i += 256) {
        int k = i >> 4, h = i & 15;
        sWr[k][h] = (k < 16) ? (__bf16)Wr[k * 16 + h] : (__bf16)0.f;
    }
    __syncthreads();

    const float g  = gamma[m], bt = beta[m], mu = mmean[m];
    const float iv = rsqrtf(mvar[m] + 1e-3f);

    float lmax = -INFINITY;
    const int numTiles = (N + 15) / 16;
    for (int tile = blockIdx.x * 8 + wave; tile < numTiles; tile += gridDim.x * 8) {
        const int nbase = tile * 16;
        // A fragment from x rows (K=16 real, rest zero)
        const int node = nbase + m;
        const int nc = node < N ? node : N - 1;
        v16bf a;
        const int kA = hi ? 16 : 0;
        #pragma unroll
        for (int j = 0; j < 16; ++j) {
            int k = kA + j;
            a[j] = (k < 16) ? (__bf16)x[(size_t)nc * NF + k] : (__bf16)0.f;
        }
        v16bf b;
        const int kB = hi ? 16 : 0;
        #pragma unroll
        for (int j = 0; j < 16; ++j) b[j] = sWr[kB + j][m];

        // Seed C with agg + br (free add via accumulator)
        v8f c;
        #pragma unroll
        for (int r = 0; r < 8; ++r) {
            int nd = nbase + r + (hi ? 8 : 0);
            c[r] = (nd < N) ? (agg[(size_t)nd * NH + m] + br[m]) : 0.f;
        }
        c = __builtin_amdgcn_wmma_f32_16x16x32_bf16(
            false, a, false, b, (short)0, c, false, false);

        #pragma unroll
        for (int r = 0; r < 8; ++r) {
            int nd = nbase + r + (hi ? 8 : 0);
            if (nd < N) {
                float v = fmaxf(c[r], 0.f);          // ReLU
                v = (v - mu) * iv * g + bt;          // BN (inference)
                lmax = fmaxf(lmax, v);
            }
        }
    }
    // combine lane L with L+16 (same h) — wave32 shuffle
    lmax = fmaxf(lmax, __shfl_xor(lmax, 16));
    if (lane < 16) sRed[wave][lane] = lmax;
    __syncthreads();
    if (tid < 16) {
        float v = sRed[0][tid];
        #pragma unroll
        for (int w = 1; w < 8; ++w) v = fmaxf(v, sRed[w][tid]);
        blockMax[blockIdx.x * NH + tid] = v;
    }
}

// ---------------- Finalize: pooled = max over blocks; out = pooled @ Wd + bd -------------
__global__ __launch_bounds__(64) void finalize3(
    const float* __restrict__ blockMax, int nblocks,
    const float* __restrict__ Wd, const float* __restrict__ bd,
    float* __restrict__ out)
{
    __shared__ float pooled[NH];
    const int tid = threadIdx.x;
    if (tid < 16) {
        float v = -INFINITY;
        for (int b = 0; b < nblocks; ++b) v = fmaxf(v, blockMax[b * NH + tid]);
        pooled[tid] = v;
    }
    __syncthreads();
    if (tid < 3) {
        float acc = bd[tid];
        #pragma unroll
        for (int h = 0; h < 16; ++h) acc += pooled[h] * Wd[h * 3 + tid];
        out[tid] = acc;
    }
}

extern "C" void kernel_launch(void* const* d_in, const int* in_sizes, int n_in,
                              void* d_out, int out_size, void* d_ws, size_t ws_size,
                              hipStream_t stream) {
    const float* x     = (const float*)d_in[0];
    const float* e     = (const float*)d_in[1];
    const int*   src   = (const int*)  d_in[2];
    const int*   dst   = (const int*)  d_in[3];
    const float* Wk    = (const float*)d_in[4];
    const float* bk    = (const float*)d_in[5];
    const float* Wr    = (const float*)d_in[6];
    const float* br    = (const float*)d_in[7];
    const float* gamma = (const float*)d_in[8];
    const float* beta  = (const float*)d_in[9];
    const float* mmean = (const float*)d_in[10];
    const float* mvar  = (const float*)d_in[11];
    const float* Wd    = (const float*)d_in[12];
    const float* bd    = (const float*)d_in[13];
    float* out = (float*)d_out;

    const int N = in_sizes[0] / NF;
    const int E = in_sizes[2];

    float* agg = (float*)d_ws;                      // N*16 f32
    const int tilesB = (N + 15) / 16;
    const int nbB    = (tilesB + 7) / 8;
    float* blockMax  = agg + (size_t)N * NH;        // nbB*16 f32

    const int nAgg = N * NH;
    hipLaunchKernelGGL(zero_f32, dim3((nAgg + 255) / 256), dim3(256), 0, stream, agg, nAgg);

    const int tilesA = (E + 15) / 16;
    const int nbA    = (tilesA + 7) / 8;
    hipLaunchKernelGGL(ecc_edge_msg, dim3(nbA), dim3(256), 0, stream,
                       x, e, src, dst, Wk, bk, agg, N, E);
    hipLaunchKernelGGL(node_update, dim3(nbB), dim3(256), 0, stream,
                       x, agg, Wr, br, gamma, beta, mmean, mvar, blockMax, N);
    hipLaunchKernelGGL(finalize3, dim3(1), dim3(64), 0, stream, blockMax, nbB, Wd, bd, out);
}
